// MambaBlockBatched_16990890622994
// MI455X (gfx1250) — compile-verified, matrix operand order checked by reading, not run on hardware
//
#include <hip/hip_runtime.h>
#include <hip/hip_bf16.h>
#include <math.h>

#define D_MODEL 512
#define D_STATE 16
#define D_CONVW 4
#define DT_RANK 32
#define BATCH   4
#define SEQ     2048
#define FN      32

typedef __attribute__((ext_vector_type(16))) __bf16 v16bf;
typedef __attribute__((ext_vector_type(8)))  float  v8f;

__device__ __forceinline__ float silu_f(float x) { return x / (1.0f + __expf(-x)); }

// ---- WMMA fragment loader --------------------------------------------------
// 16-bit A 16x32 layout (ISA 7.12.2): lanes 0-15 hold row = lane&15; khalf=lane>>4.
// elements 0..7  -> K = 8*khalf + e
// elements 8..15 -> K = 16 + 8*khalf + (e-8)
// The B fragment of W[N,K] row-major (so B[k,n] = W[n,k]) uses the identical
// gather with row = N index: exactly what X @ W^T needs. All GEMM operands are
// arranged row-major-over-K so this is the only loader (contiguous float4s).
__device__ __forceinline__ v16bf load_frag_rowK(const float* __restrict__ rowp, int khalf) {
  const float4* q0 = reinterpret_cast<const float4*>(rowp + 8 * khalf);
  const float4* q1 = reinterpret_cast<const float4*>(rowp + 16 + 8 * khalf);
  float4 a0 = q0[0], a1 = q0[1], b0 = q1[0], b1 = q1[1];
  v16bf f;
  f[0]=(__bf16)a0.x;  f[1]=(__bf16)a0.y;  f[2]=(__bf16)a0.z;  f[3]=(__bf16)a0.w;
  f[4]=(__bf16)a1.x;  f[5]=(__bf16)a1.y;  f[6]=(__bf16)a1.z;  f[7]=(__bf16)a1.w;
  f[8]=(__bf16)b0.x;  f[9]=(__bf16)b0.y;  f[10]=(__bf16)b0.z; f[11]=(__bf16)b0.w;
  f[12]=(__bf16)b1.x; f[13]=(__bf16)b1.y; f[14]=(__bf16)b1.z; f[15]=(__bf16)b1.w;
  return f;
}

// ---- bf16 WMMA GEMM with 2x2 register blocking -----------------------------
// C[M,N] = A[M,K] @ W[N,K]^T, f32 accumulate. Each wave owns a 32x32 tile:
// 4 accumulators, 4 WMMAs per 32-wide K step (A/B fragments reused 2x each).
// MODE: 0 = plain store
//       2 = finite-guard store (out_proj)
//       3 = +bias[n], softplus, store batch-transposed: [b][n][l] (dt_proj)
template <int MODE>
__global__ void gemm_bf16_wmma(const float* __restrict__ A, int lda, long long strideA,
                               const float* __restrict__ W, int ldw, long long strideW,
                               float* __restrict__ C, int ldc, long long strideC,
                               const float* __restrict__ bias,
                               int M, int N, int K) {
  const int batch = blockIdx.z;
  A += (size_t)batch * strideA;
  W += (size_t)batch * strideW;
  C += (size_t)batch * strideC;

  const int lane   = threadIdx.x & 31;
  const int wid    = threadIdx.x >> 5;
  const int tilesN = N >> 5;                       // 32-wide tiles
  const int tilesM = M >> 5;
  const int tid    = blockIdx.x * (blockDim.x >> 5) + wid;
  if (tid >= tilesM * tilesN) return;              // uniform per wave: EXEC stays all-ones
  const int tm = tid / tilesN;
  const int tn = tid % tilesN;

  const int r16   = lane & 15;
  const int khalf = lane >> 4;
  const float* arow0 = A + (size_t)((tm << 5) + r16) * lda;
  const float* arow1 = arow0 + (size_t)16 * lda;
  const float* brow0 = W + (size_t)((tn << 5) + r16) * ldw;
  const float* brow1 = brow0 + (size_t)16 * ldw;

  v8f acc00, acc01, acc10, acc11;
#pragma unroll
  for (int i = 0; i < 8; ++i) { acc00[i] = 0.f; acc01[i] = 0.f; acc10[i] = 0.f; acc11[i] = 0.f; }

#pragma unroll 2
  for (int k0 = 0; k0 < K; k0 += 32) {
    v16bf a0 = load_frag_rowK(arow0 + k0, khalf);
    v16bf a1 = load_frag_rowK(arow1 + k0, khalf);
    v16bf b0 = load_frag_rowK(brow0 + k0, khalf);
    v16bf b1 = load_frag_rowK(brow1 + k0, khalf);
    acc00 = __builtin_amdgcn_wmma_f32_16x16x32_bf16(false, a0, false, b0, (short)0, acc00, false, false);
    acc01 = __builtin_amdgcn_wmma_f32_16x16x32_bf16(false, a0, false, b1, (short)0, acc01, false, false);
    acc10 = __builtin_amdgcn_wmma_f32_16x16x32_bf16(false, a1, false, b0, (short)0, acc10, false, false);
    acc11 = __builtin_amdgcn_wmma_f32_16x16x32_bf16(false, a1, false, b1, (short)0, acc11, false, false);
  }

  // C/D layout: lane -> column n16 = r16; VGPR v -> row v + 8*khalf.
  const int mbase = (tm << 5) + 8 * khalf;
  const int nbase = (tn << 5) + r16;
#pragma unroll
  for (int tj = 0; tj < 2; ++tj) {
    const int n  = nbase + 16 * tj;
    const float bn = (MODE == 3) ? bias[n] : 0.0f;
#pragma unroll
    for (int ti = 0; ti < 2; ++ti) {
      v8f acc = (ti == 0) ? (tj == 0 ? acc00 : acc01) : (tj == 0 ? acc10 : acc11);
#pragma unroll
      for (int v = 0; v < 8; ++v) {
        const int m = mbase + 16 * ti + v;
        float val = acc[v];
        if (MODE == 2) {
          if (!__builtin_isfinite(val)) val = 0.0f;
        } else if (MODE == 3) {
          val += bn;
          val = (val > 20.0f) ? val : log1pf(__expf(val));          // softplus
        }
        if (MODE == 3) {
          // store delta transposed per batch: [b][n][l], b = m/SEQ, l = m%SEQ
          const int bb = m >> 11;                                   // SEQ == 2048
          const int ll = m & (SEQ - 1);
          C[((size_t)bb * D_MODEL + n) * SEQ + ll] = val;
        } else {
          C[(size_t)m * ldc + n] = val;
        }
      }
    }
  }
}

// ---- FiLM: gamma/beta per batch (tiny) -------------------------------------
__global__ void film_kernel(const float* __restrict__ gpt,
                            const float* __restrict__ w1, const float* __restrict__ b1,
                            const float* __restrict__ w2, const float* __restrict__ b2,
                            float* __restrict__ gb) {
  const int b = blockIdx.x;
  __shared__ float gs[D_MODEL];
  __shared__ float hs[D_MODEL];
  for (int i = threadIdx.x; i < D_MODEL; i += blockDim.x) gs[i] = gpt[b * D_MODEL + i];
  __syncthreads();
  for (int j = threadIdx.x; j < D_MODEL; j += blockDim.x) {
    const float* wr = w1 + (size_t)j * D_MODEL;
    float s = b1[j];
    for (int k = 0; k < D_MODEL; ++k) s += gs[k] * wr[k];
    hs[j] = silu_f(s);
  }
  __syncthreads();
  for (int j = threadIdx.x; j < 2 * D_MODEL; j += blockDim.x) {
    const float* wr = w2 + (size_t)j * D_MODEL;
    float s = b2[j];
    for (int k = 0; k < D_MODEL; ++k) s += hs[k] * wr[k];
    gb[b * 2 * D_MODEL + j] = s;
  }
}

// ---- Cross-attention (F=32) + residual + FiLM modulation -------------------
// One wave per (b,l): lane f computes score f, softmax across the wave,
// then all lanes cooperatively produce the 512-dim output row.
__global__ void attn_film_kernel(const float* __restrict__ atoms,
                                 const float* __restrict__ fg,
                                 const float* __restrict__ gb,
                                 float* __restrict__ xout) {
  const int b    = blockIdx.y;
  const int wid  = threadIdx.x >> 5;
  const int lane = threadIdx.x & 31;
  const int l    = blockIdx.x * 8 + wid;
  __shared__ float s_attn[8][FN];

  const float* arow = atoms + ((size_t)b * SEQ + l) * D_MODEL;
  const float* fgb  = fg + (size_t)b * FN * D_MODEL;

  const float4* ap = reinterpret_cast<const float4*>(arow);
  const float4* fp = reinterpret_cast<const float4*>(fgb + lane * D_MODEL);
  float s = 0.0f;
#pragma unroll 4
  for (int i = 0; i < D_MODEL / 4; ++i) {
    float4 a = ap[i], f = fp[i];
    s += a.x * f.x + a.y * f.y + a.z * f.z + a.w * f.w;
  }
  s *= 0.044194173824159216f;  // 1/sqrt(512)

  float mx = s;
#pragma unroll
  for (int off = 16; off; off >>= 1) mx = fmaxf(mx, __shfl_xor(mx, off, 32));
  float e = __expf(s - mx);
  float sum = e;
#pragma unroll
  for (int off = 16; off; off >>= 1) sum += __shfl_xor(sum, off, 32);
  s_attn[wid][lane] = e / sum;
  __syncthreads();

  const float* gbp = gb + b * 2 * D_MODEL;
  float* orow = xout + ((size_t)b * SEQ + l) * D_MODEL;
  for (int d = lane; d < D_MODEL; d += 32) {
    float acc = arow[d];
#pragma unroll 8
    for (int f = 0; f < FN; ++f) acc += s_attn[wid][f] * fgb[(size_t)f * D_MODEL + d];
    orow[d] = acc * gbp[d] + gbp[D_MODEL + d];
  }
}

// ---- Depthwise causal conv (W=4) + SiLU ------------------------------------
// xs is cols [0,512) of the in_proj output laid out [B*SEQ, 1024].
__global__ void conv_silu_kernel(const float* __restrict__ inproj,
                                 const float* __restrict__ cw,
                                 const float* __restrict__ cb,
                                 float* __restrict__ u) {
  const int idx = blockIdx.x * blockDim.x + threadIdx.x;
  const int d = idx & (D_MODEL - 1);
  const int l = (idx >> 9) & (SEQ - 1);
  const int b = idx >> 20;  // 9 + 11
  float s = cb[d];
#pragma unroll
  for (int w = 0; w < D_CONVW; ++w) {
    int ls = l - (D_CONVW - 1) + w;
    if (ls >= 0)
      s += cw[d * D_CONVW + w] * inproj[((size_t)b * SEQ + ls) * 1024 + d];
  }
  u[((size_t)b * SEQ + l) * D_MODEL + d] = silu_f(s);
}

// ---- Fused SSM scan + skip(D) + gate(silu(res)) ----------------------------
// One thread per (b,d) channel; 16-state carry in VGPRs; sequential over L.
// Avoids materializing deltaA / deltaB_u (2 x 268 MB in the reference).
__global__ void scan_kernel(const float* __restrict__ delta_p,
                            const float* __restrict__ u,
                            const float* __restrict__ xdbl,     // [M,64]: [.,32:48]=B, [.,48:64]=C
                            const float* __restrict__ inproj,   // res = cols [512,1024)
                            const float* __restrict__ A_log,
                            const float* __restrict__ Dp,
                            float* __restrict__ yg) {
  const int b = blockIdx.y;
  const int d = blockIdx.x * blockDim.x + threadIdx.x;

  float Ac[D_STATE], carry[D_STATE];
#pragma unroll
  for (int j = 0; j < D_STATE; ++j) {
    Ac[j] = -__expf(A_log[d * D_STATE + j]);
    carry[j] = 0.0f;
  }
  const float Dd = Dp[d];

  for (int l = 0; l < SEQ; ++l) {
    const size_t md = (size_t)b * SEQ + l;
    const float dp = delta_p[md * D_MODEL + d];
    const float uu = u[md * D_MODEL + d];
    float Bv[D_STATE], Cv[D_STATE];
    const float4* bc = reinterpret_cast<const float4*>(xdbl + md * 64 + DT_RANK);
#pragma unroll
    for (int i = 0; i < 4; ++i) { reinterpret_cast<float4*>(Bv)[i] = bc[i];
                                  reinterpret_cast<float4*>(Cv)[i] = bc[4 + i]; }
    float y = uu * Dd;
    const float dbu = dp * uu;
#pragma unroll
    for (int j = 0; j < D_STATE; ++j) {
      carry[j] = __expf(dp * Ac[j]) * carry[j] + dbu * Bv[j];
      y += carry[j] * Cv[j];
    }
    const float r = inproj[md * 1024 + D_MODEL + d];
    yg[md * D_MODEL + d] = y * silu_f(r);
  }
}

// ---- Host-side launcher ----------------------------------------------------
static void launch_gemm(hipStream_t stream, int mode,
                        const float* A, int lda, long long sA,
                        const float* W, int ldw, long long sW,
                        float* C, int ldc, long long sC,
                        const float* bias, int M, int N, int K, int batches) {
  int tiles = (M >> 5) * (N >> 5);                 // 32x32 tiles, one per wave
  dim3 grid((tiles + 7) / 8, 1, batches);
  switch (mode) {
    case 0: gemm_bf16_wmma<0><<<grid, 256, 0, stream>>>(A, lda, sA, W, ldw, sW, C, ldc, sC, bias, M, N, K); break;
    case 2: gemm_bf16_wmma<2><<<grid, 256, 0, stream>>>(A, lda, sA, W, ldw, sW, C, ldc, sC, bias, M, N, K); break;
    default: gemm_bf16_wmma<3><<<grid, 256, 0, stream>>>(A, lda, sA, W, ldw, sW, C, ldc, sC, bias, M, N, K); break;
  }
}

extern "C" void kernel_launch(void* const* d_in, const int* in_sizes, int n_in,
                              void* d_out, int out_size, void* d_ws, size_t ws_size,
                              hipStream_t stream) {
  (void)in_sizes; (void)n_in; (void)out_size; (void)ws_size;
  const float* atoms      = (const float*)d_in[0];
  const float* dis        = (const float*)d_in[1];
  const float* gpt        = (const float*)d_in[2];
  const float* fg         = (const float*)d_in[3];
  const float* in_proj_w  = (const float*)d_in[4];
  const float* conv_w     = (const float*)d_in[5];
  const float* conv_b     = (const float*)d_in[6];
  const float* x_proj_w   = (const float*)d_in[7];
  const float* dt_proj_w  = (const float*)d_in[8];
  const float* dt_proj_b  = (const float*)d_in[9];
  const float* A_log      = (const float*)d_in[10];
  const float* D_param    = (const float*)d_in[11];
  const float* out_proj_w = (const float*)d_in[12];
  const float* film_w1    = (const float*)d_in[13];
  const float* film_b1    = (const float*)d_in[14];
  const float* film_w2    = (const float*)d_in[15];
  const float* film_b2    = (const float*)d_in[16];
  float* out = (float*)d_out;

  const size_t M = (size_t)BATCH * SEQ;           // 8192
  float* ws     = (float*)d_ws;
  float* gb     = ws;                              // [B,1024]
  float* xbuf   = gb + 4096;                       // [M,512]   fused input
  float* inproj = xbuf + M * 512;                  // [M,1024]  xs|res
  float* ubuf   = inproj + M * 1024;               // [M,512]
  float* xdbl   = ubuf + M * 512;                  // [M,64]    delta_r|B|C
  float* deltaT = xdbl + M * 64;                   // [B][512][2048] transposed delta
  float* dp     = deltaT + M * 512;                // [M,512]
  float* yg     = xbuf;                            // reuse (dead after in_proj)

  // 1) FiLM coefficients
  film_kernel<<<BATCH, 256, 0, stream>>>(gpt, film_w1, film_b1, film_w2, film_b2, gb);
  // 2) cross-attention + residual + FiLM
  attn_film_kernel<<<dim3(SEQ / 8, BATCH), 256, 0, stream>>>(atoms, fg, gb, xbuf);
  // 3) in_proj: [8192,1024] = x @ W^T  (WMMA)
  launch_gemm(stream, 0, xbuf, 512, 0, in_proj_w, 512, 0, inproj, 1024, 0,
              nullptr, (int)M, 1024, 512, 1);
  // 4) depthwise causal conv + SiLU
  conv_silu_kernel<<<(BATCH * SEQ * D_MODEL) / 256, 256, 0, stream>>>(inproj, conv_w, conv_b, ubuf);
  // 5) x_proj: [8192,64] = u @ W^T  (WMMA)
  launch_gemm(stream, 0, ubuf, 512, 0, x_proj_w, 512, 0, xdbl, 64, 0,
              nullptr, (int)M, 64, 512, 1);
  // 6) dt_proj + softplus, stored TRANSPOSED per batch -> deltaT[b][d][l]  (WMMA, K=32)
  launch_gemm(stream, 3, xdbl, 64, 0, dt_proj_w, 32, 0, deltaT, 0, 0,
              dt_proj_b, (int)M, 512, 32, 1);
  // 7) graph mix (largest GEMM): dp[b] = dis[b] @ delta[b] == dis[b] @ deltaT[b]^T
  //    -> both operands row-major-over-K, contiguous fragment loads (batched WMMA)
  launch_gemm(stream, 0, dis, SEQ, (long long)SEQ * SEQ,
              deltaT, SEQ, (long long)D_MODEL * SEQ,
              dp, D_MODEL, (long long)SEQ * D_MODEL, nullptr, SEQ, D_MODEL, SEQ, BATCH);
  // 8) fused SSM scan + D-skip + silu(res) gate
  scan_kernel<<<dim3(D_MODEL / 256, BATCH), 256, 0, stream>>>(dp, ubuf, xdbl, inproj,
                                                              A_log, D_param, yg);
  // 9) out_proj with finite-guard epilogue  (WMMA)
  launch_gemm(stream, 2, yg, 512, 0, out_proj_w, 512, 0, out, 512, 0,
              nullptr, (int)M, 512, 512, 1);
}
